// GATLayer_23708219474193
// MI455X (gfx1250) — matrix operand order repaired and checked by reading, not compiled, not run on hardware
//
#include <hip/hip_runtime.h>

#define BATCH 16
#define NNODE 1024
#define INDIM 128
#define NHEAD 4
#define HDIM  64
#define LOG2E 1.4426950408889634f

typedef __bf16 v16bf __attribute__((ext_vector_type(16)));
typedef float  v8f   __attribute__((ext_vector_type(8)));

union BF16x16 {
  v16bf v;
  unsigned short u[16];
  uint4 q[2];
};
union U16x8 {
  uint4 q;
  unsigned short u[8];
};

__device__ __forceinline__ unsigned short f32_bf16(float f) {
  __bf16 b = (__bf16)f;                     // native v_cvt on gfx1250
  return __builtin_bit_cast(unsigned short, b);
}

__device__ __forceinline__ float fast_exp2(float x) {
  return __builtin_amdgcn_exp2f(x);         // single v_exp_f32
}

// ---------------------------------------------------------------------------
// Kernel 0: one-time swizzle of W[h,k,d] (f32) into bf16 WMMA B-operand order.
// ---------------------------------------------------------------------------
__global__ __launch_bounds__(256) void w_swizzle_kernel(
    const float* __restrict__ W, unsigned short* __restrict__ Wsw)
{
  const int i = blockIdx.x * 256 + threadIdx.x;   // 0 .. 32767
  const int e    = i & 15;
  const int lane = (i >> 4) & 31;
  const int ot   = (i >> 9) & 3;
  const int kc   = (i >> 11) & 3;
  const int head = i >> 13;
  const int k   = kc * 32 + (e >> 3) * 16 + (lane >> 4) * 8 + (e & 7);
  const int col = ot * 16 + (lane & 15);
  Wsw[i] = f32_bf16(W[((size_t)head * INDIM + k) * HDIM + col]);
}

// ---------------------------------------------------------------------------
// Kernel 1: Wh = h @ W (bf16 WMMA, f32 acc); f1 = Wh.a1, f2 = Wh.a2.
// Wh is written pre-swizzled into WMMA B-operand order for kernel 2.
// ---------------------------------------------------------------------------
__global__ __launch_bounds__(128) void gat_wh_kernel(
    const float* __restrict__ h, const unsigned short* __restrict__ Wsw,
    const float* __restrict__ a,
    unsigned short* __restrict__ Whb,
    float* __restrict__ f1, float* __restrict__ f2)
{
  const int wave = threadIdx.x >> 5;
  const int lane = threadIdx.x & 31;
  const int tile = blockIdx.x * 4 + wave;        // 0 .. 4095
  const int b    = tile >> 8;
  const int rem  = tile & 255;
  const int head = rem >> 6;
  const int n0   = (rem & 63) << 4;
  const int mrow = lane & 15;
  const int half = lane >> 4;

  BF16x16 A[4];
  const float* hrow = h + ((size_t)b * NNODE + (n0 + mrow)) * INDIM;
#pragma unroll
  for (int kc = 0; kc < 4; ++kc) {
    const float4* s0 = (const float4*)(hrow + kc * 32 + half * 8);
    const float4* s1 = (const float4*)(hrow + kc * 32 + 16 + half * 8);
    const float4 x0 = s0[0], x1 = s0[1], y0 = s1[0], y1 = s1[1];
    const float xs[16] = {x0.x, x0.y, x0.z, x0.w, x1.x, x1.y, x1.z, x1.w,
                          y0.x, y0.y, y0.z, y0.w, y1.x, y1.y, y1.z, y1.w};
#pragma unroll
    for (int e = 0; e < 16; ++e) A[kc].u[e] = f32_bf16(xs[e]);
  }

  v8f acc[4];
#pragma unroll
  for (int ot = 0; ot < 4; ++ot)
#pragma unroll
    for (int r = 0; r < 8; ++r) acc[ot][r] = 0.0f;

  const uint4* WB = (const uint4*)Wsw;
#pragma unroll
  for (int ot = 0; ot < 4; ++ot) {
#pragma unroll
    for (int kc = 0; kc < 4; ++kc) {
      BF16x16 Bm;
      const int vidx = ((((head * 4 + kc) * 4 + ot) * 32) + lane) * 2;
      Bm.q[0] = WB[vidx];
      Bm.q[1] = WB[vidx + 1];
      acc[ot] = __builtin_amdgcn_wmma_f32_16x16x32_bf16(
          false, A[kc].v, false, Bm.v, (short)0, acc[ot], false, false);
    }
  }

  const size_t bh = (size_t)b * NHEAD + head;
  const int mc     = n0 >> 5;
  const int half_d = (n0 >> 4) & 1;
  float p1[8], p2[8];
#pragma unroll
  for (int r = 0; r < 8; ++r) { p1[r] = 0.0f; p2[r] = 0.0f; }
#pragma unroll
  for (int ot = 0; ot < 4; ++ot) {
    const int d  = ot * 16 + mrow;
    const float a1 = a[head * 2 * HDIM + d];
    const float a2 = a[head * 2 * HDIM + HDIM + d];
    U16x8 pk;
#pragma unroll
    for (int r = 0; r < 8; ++r) {
      pk.u[r] = f32_bf16(acc[ot][r]);
      p1[r] += acc[ot][r] * a1;
      p2[r] += acc[ot][r] * a2;
    }
    const size_t base =
        (((bh * 32 + mc) * 4 + ot) * 32 + half * 16 + mrow) * 16 + half_d * 8;
    *(uint4*)(Whb + base) = pk.q;
  }
#pragma unroll
  for (int r = 0; r < 8; ++r) {
#pragma unroll
    for (int m = 8; m >= 1; m >>= 1) {
      p1[r] += __shfl_xor(p1[r], m, 32);
      p2[r] += __shfl_xor(p2[r], m, 32);
    }
  }
  if (mrow == 0) {
#pragma unroll
    for (int r = 0; r < 8; ++r) {
      const int n = n0 + half * 8 + r;
      f1[bh * NNODE + n] = p1[r];
      f2[bh * NNODE + n] = p2[r];
    }
  }
}

// ---------------------------------------------------------------------------
// Kernel 2: fused masked-softmax attention + P@Wh + ELU (base-2 softmax).
// Block = (b, 16-row tile); 4 waves = 4 heads share one adj bitmask in LDS.
// ---------------------------------------------------------------------------
__global__ __launch_bounds__(128) void gat_attn_kernel(
    const int* __restrict__ adj, const unsigned short* __restrict__ Whb,
    const float* __restrict__ f1, const float* __restrict__ f2,
    float* __restrict__ out)
{
  __shared__ __align__(16) unsigned int adjbits[16 * 32]; // 16x1024 bits
  __shared__ __align__(16) float f2s[NHEAD * NNODE];      // 16 KB (pre-scaled)

  const int wave = threadIdx.x >> 5;          // = head
  const int lane = threadIdx.x & 31;
  const int b    = blockIdx.x >> 6;
  const int n0   = (blockIdx.x & 63) << 4;
  const int mrow = lane & 15;
  const int half = lane >> 4;

  // Bitmask build: thread handles 4 words (128 cols) of one row, int4 loads.
  {
    const int row = threadIdx.x >> 3;
    const int c0  = (threadIdx.x & 7) << 7;
    const int4* ap = (const int4*)(adj + (((size_t)b * NNODE + n0 + row) * NNODE + c0));
#pragma unroll
    for (int w = 0; w < 4; ++w) {
      unsigned int bits = 0;
#pragma unroll
      for (int j = 0; j < 8; ++j) {
        const int4 v = ap[w * 8 + j];   // adj values are 0/1
        bits |= ((unsigned)v.x & 1u) << (j * 4 + 0);
        bits |= ((unsigned)v.y & 1u) << (j * 4 + 1);
        bits |= ((unsigned)v.z & 1u) << (j * 4 + 2);
        bits |= ((unsigned)v.w & 1u) << (j * 4 + 3);
      }
      adjbits[(row << 5) + (c0 >> 5) + w] = bits;
    }
  }

  // Stage f2 pre-scaled by log2(e): leaky_relu commutes with positive scale,
  // so the whole softmax runs in base-2 (v_exp_f32 is native exp2).
  const size_t bh = (size_t)b * NHEAD + wave;
  for (int i = lane; i < NNODE; i += 32)
    f2s[wave * NNODE + i] = f2[bh * NNODE + i] * LOG2E;

  const float f1v = f1[bh * NNODE + n0 + mrow] * LOG2E;
  __syncthreads();

  v8f acc[4];
#pragma unroll
  for (int ot = 0; ot < 4; ++ot)
#pragma unroll
    for (int r = 0; r < 8; ++r) acc[ot][r] = 0.0f;

  float M = -9.0e15f, S = 0.0f;
  const uint4* WhbV = (const uint4*)Whb + bh * 32 * 4 * 32 * 2;

#pragma unroll 1
  for (int mc = 0; mc < 32; ++mc) {
    const int m0 = mc << 5;
    const unsigned int aw = adjbits[(mrow << 5) + mc];

    const float4* fz0 = (const float4*)&f2s[wave * NNODE + m0 + half * 8];
    const float4* fz1 = (const float4*)&f2s[wave * NNODE + m0 + 16 + half * 8];
    const float4 a0 = fz0[0], a1v = fz0[1], b0 = fz1[0], b1 = fz1[1];
    const float fv[16] = {a0.x, a0.y, a0.z, a0.w, a1v.x, a1v.y, a1v.z, a1v.w,
                          b0.x, b0.y, b0.z, b0.w, b1.x,  b1.y,  b1.z,  b1.w};

    float ev[16];
    float cmax = -9.0e15f;
#pragma unroll
    for (int e = 0; e < 16; ++e) {
      const int bitpos = ((e >> 3) << 4) + half * 8 + (e & 7); // col & 31
      float x = f1v + fv[e];
      x = fmaxf(x, 0.2f * x);                        // leaky_relu(0.2)
      const bool on = (aw >> bitpos) & 1u;
      ev[e] = on ? x : -9.0e15f;
      cmax = fmaxf(cmax, ev[e]);
    }
    cmax = fmaxf(cmax, __shfl_xor(cmax, 16, 32));
    const float newM = fmaxf(M, cmax);

    // exp2(masked - newM) underflows to exactly 0; no second select needed.
    float psum = 0.0f;
    BF16x16 P;
#pragma unroll
    for (int e = 0; e < 16; ++e) {
      const float pe = fast_exp2(ev[e] - newM);
      psum += pe;
      P.u[e] = f32_bf16(pe);
    }
    psum += __shfl_xor(psum, 16, 32);

    if (__any(cmax > M)) {                 // some row's max grew: rescale
      const float scale = fast_exp2(M - newM);
      S = S * scale + psum;
      M = newM;
      float srow[8];
#pragma unroll
      for (int r = 0; r < 8; ++r) srow[r] = __shfl(scale, half * 8 + r, 32);
#pragma unroll
      for (int ot = 0; ot < 4; ++ot)
#pragma unroll
        for (int r = 0; r < 8; ++r) acc[ot][r] *= srow[r];
    } else {
      S += psum;                           // common fast path
    }

#pragma unroll
    for (int ot = 0; ot < 4; ++ot) {
      BF16x16 Bm;                          // pre-swizzled Wh: 2 coalesced b128
      const int vidx = ((mc * 4 + ot) * 32 + lane) * 2;
      Bm.q[0] = WhbV[vidx];
      Bm.q[1] = WhbV[vidx + 1];
      acc[ot] = __builtin_amdgcn_wmma_f32_16x16x32_bf16(
          false, P.v, false, Bm.v, (short)0, acc[ot], false, false);
    }
  }

  // Finalize: divide by row sum, ELU, write transposed (b, n, h*64+d).
  const float rinv = (S > 0.0f) ? (1.0f / S) : 0.0f;
  float rrow[8];
#pragma unroll
  for (int r = 0; r < 8; ++r) rrow[r] = __shfl(rinv, half * 8 + r, 32);

#pragma unroll
  for (int ot = 0; ot < 4; ++ot) {
#pragma unroll
    for (int r = 0; r < 8; ++r) {
      float v = acc[ot][r] * rrow[r];
      v = (v > 0.0f) ? v : (__expf(v) - 1.0f);
      const int n = n0 + half * 8 + r;
      out[((size_t)b * NNODE + n) * (NHEAD * HDIM) + wave * HDIM + ot * 16 + mrow] = v;
    }
  }
}

// ---------------------------------------------------------------------------
extern "C" void kernel_launch(void* const* d_in, const int* in_sizes, int n_in,
                              void* d_out, int out_size, void* d_ws, size_t ws_size,
                              hipStream_t stream) {
  const float* h   = (const float*)d_in[0];
  const int*   adj = (const int*)d_in[1];
  const float* W   = (const float*)d_in[2];
  const float* a   = (const float*)d_in[3];
  float* out = (float*)d_out;

  unsigned short* Whb = (unsigned short*)d_ws;                   // 8 MB (swizzled)
  const size_t whbElems = (size_t)BATCH * NHEAD * NNODE * HDIM;
  float* f1 = (float*)((char*)d_ws + whbElems * sizeof(unsigned short));
  float* f2 = f1 + (size_t)BATCH * NHEAD * NNODE;
  unsigned short* Wsw = (unsigned short*)(f2 + (size_t)BATCH * NHEAD * NNODE); // 64 KB

  w_swizzle_kernel<<<(NHEAD * INDIM * HDIM) / 256, 256, 0, stream>>>(W, Wsw);
  gat_wh_kernel<<<BATCH * NHEAD * (NNODE / 16) / 4, 128, 0, stream>>>(
      h, Wsw, a, Whb, f1, f2);
  gat_attn_kernel<<<BATCH * (NNODE / 16), 128, 0, stream>>>(
      adj, Whb, f1, f2, out);
}
